// GNN_34772055228550
// MI455X (gfx1250) — matrix operand order repaired and checked by reading, not compiled, not run on hardware
//
#include <hip/hip_runtime.h>
#include <hip/hip_bf16.h>

typedef __attribute__((ext_vector_type(16))) _Float16 v16h;
typedef __attribute__((ext_vector_type(8)))  float    v8f;

#define IN_CH  14
#define HID    16
#define OUT_CH 2

// ---------------------------------------------------------------------------
// 0) zero the [deg | agg1] prefix of the workspace (ws not re-poisoned between
//    replays, so we must re-zero every call)
// ---------------------------------------------------------------------------
__global__ void k_zero(float* __restrict__ p, size_t n) {
    size_t i = (size_t)blockIdx.x * blockDim.x + threadIdx.x;
    if (i < n) p[i] = 0.0f;
}

// ---------------------------------------------------------------------------
// 1) degree on dst (u32 atomics), then deg -> rsqrt(deg + 1) in place
// ---------------------------------------------------------------------------
__global__ void k_degree(const long long* __restrict__ dst,
                         unsigned* __restrict__ deg, int nE) {
    int e = blockIdx.x * blockDim.x + threadIdx.x;
    if (e < nE) atomicAdd(&deg[(int)dst[e]], 1u);
}

__global__ void k_dinv(unsigned* __restrict__ deg, int n) {
    int i = blockIdx.x * blockDim.x + threadIdx.x;
    if (i < n) {
        float d = (float)deg[i] + 1.0f;            // +1 self loop
        deg[i] = __float_as_uint(rsqrtf(d));       // reinterpret slot as f32
    }
}

// ---------------------------------------------------------------------------
// 2) h1 = x @ W1 via v_wmma_f32_16x16x32_f16; one wave owns a 16-node tile.
//    A (16x32 f16): lane L holds row M=L&15; elements 0..7 = K in
//      [8*(L>=16), +8); elements 8..15 = K>=16 -> zero (IN_CH=14).
//    B (32x16 f16): lane L holds column N=L&15; element e<8 = K 8*(L>=16)+e, e>=8 = K 16+8*(L>=16)+e-8.
//    C/D (16x16 f32): VGPR r -> M = r + 8*(L>=16), N = L&15.
//    Padding is branch-free: unconditional load at a clamped (in-bounds)
//    index + VALU select to zero, so no exec-mask ladders are emitted.
// ---------------------------------------------------------------------------
__global__ __launch_bounds__(256) void k_gemm1_wmma(
        const float* __restrict__ x, const float* __restrict__ W1,
        float* __restrict__ h1, int n_nodes) {
    const int lane = threadIdx.x & 31;
    const int wave = blockIdx.x * (blockDim.x >> 5) + (threadIdx.x >> 5);
    const int row0 = wave * 16;
    if (row0 >= n_nodes) return;                   // wave-uniform: EXEC stays all-1s

    const int m  = lane & 15;
    const int hi = lane >> 4;                      // half-wave select
    const int row = min(row0 + m, n_nodes - 1);
    const int kb  = hi * 8;

    v16h a;
    #pragma unroll
    for (int e = 0; e < 8; ++e) {
        int k = kb + e;
        float v = x[(size_t)row * IN_CH + min(k, IN_CH - 1)]; // always in-bounds
        v = (k < IN_CH) ? v : 0.0f;                           // v_cndmask, no branch
        a[e]     = (_Float16)v;
        a[8 + e] = (_Float16)0.0f;                 // K >= 16 padding
    }
    v16h b;
    #pragma unroll
    for (int e = 0; e < 16; ++e) {
        int k = (e < 8) ? hi * 8 + e : 16 + hi * 8 + (e - 8);
        float v = W1[min(k, IN_CH - 1) * HID + m];
        v = (k < IN_CH) ? v : 0.0f;
        b[e] = (_Float16)v;
    }
    v8f c = {};
    c = __builtin_amdgcn_wmma_f32_16x16x32_f16(
            /*neg_a=*/false, a, /*neg_b=*/false, b,
            /*c_mod=*/(short)0, c, /*reuse_a=*/false, /*reuse_b=*/false);

    if (row0 + 16 <= n_nodes) {                    // full tile: wave-uniform fast path
        #pragma unroll
        for (int r = 0; r < 8; ++r) {
            int orow = row0 + r + hi * 8;
            h1[(size_t)orow * HID + m] = c[r];     // unconditional stores
        }
    } else {                                       // partial tile (unused for N%16==0)
        #pragma unroll
        for (int r = 0; r < 8; ++r) {
            int orow = row0 + r + hi * 8;
            if (orow < n_nodes) h1[(size_t)orow * HID + m] = c[r];
        }
    }
}

// ---------------------------------------------------------------------------
// 3) layer-1 edge aggregation: 4 lanes per edge, float4 gather + 4 f32 atomics
// ---------------------------------------------------------------------------
__global__ void k_agg1(const long long* __restrict__ src,
                       const long long* __restrict__ dst,
                       const float* __restrict__ dinv,
                       const float* __restrict__ h1,
                       float* __restrict__ agg1, int nE) {
    long long t = (long long)blockIdx.x * blockDim.x + threadIdx.x;
    int e = (int)(t >> 2);
    if (e >= nE) return;
    int ep = e + 8192;                             // stream-ahead on index array
    if (ep < nE) {
        __builtin_prefetch(&src[ep], 0, 0);
        __builtin_prefetch(&dst[ep], 0, 0);
    }
    int fq = ((int)t & 3) * 4;
    int s = (int)src[e], d = (int)dst[e];
    float norm = dinv[s] * dinv[d];
    const float4 v = *(const float4*)(h1 + (size_t)s * HID + fq);
    float* o = agg1 + (size_t)d * HID + fq;
    atomicAdd(o + 0, v.x * norm);
    atomicAdd(o + 1, v.y * norm);
    atomicAdd(o + 2, v.z * norm);
    atomicAdd(o + 3, v.w * norm);
}

// ---------------------------------------------------------------------------
// 4) fused: hr = relu(agg1 + xW1*dinv^2 + b1);  h2 = hr @ W2;
//    out   = h2*dinv^2 + b2   (initializes d_out before layer-2 atomics)
// ---------------------------------------------------------------------------
__global__ void k_finish1_gemm2(const float* __restrict__ agg1,
                                const float* __restrict__ xw1,
                                const float* __restrict__ dinv,
                                const float* __restrict__ b1,
                                const float* __restrict__ W2,
                                const float* __restrict__ b2,
                                float* __restrict__ h2,
                                float* __restrict__ out, int n) {
    __shared__ float sW2[HID * OUT_CH];
    __shared__ float sB1[HID];
    if (threadIdx.x < HID * OUT_CH) sW2[threadIdx.x] = W2[threadIdx.x];
    if (threadIdx.x < HID)          sB1[threadIdx.x] = b1[threadIdx.x];
    __syncthreads();
    int i = blockIdx.x * blockDim.x + threadIdx.x;
    if (i >= n) return;
    float di  = dinv[i];
    float dii = di * di;
    float o0 = 0.0f, o1 = 0.0f;
    #pragma unroll
    for (int f = 0; f < HID; ++f) {
        float hv = fmaf(xw1[(size_t)i * HID + f], dii,
                        agg1[(size_t)i * HID + f]) + sB1[f];
        hv = fmaxf(hv, 0.0f);                      // relu
        o0 = fmaf(hv, sW2[f * OUT_CH + 0], o0);
        o1 = fmaf(hv, sW2[f * OUT_CH + 1], o1);
    }
    h2[(size_t)i * OUT_CH + 0] = o0;
    h2[(size_t)i * OUT_CH + 1] = o1;
    out[(size_t)i * OUT_CH + 0] = fmaf(o0, dii, b2[0]);
    out[(size_t)i * OUT_CH + 1] = fmaf(o1, dii, b2[1]);
}

// ---------------------------------------------------------------------------
// 5) layer-2 edge aggregation: 1 lane per edge, float2 gather + 2 f32 atomics
// ---------------------------------------------------------------------------
__global__ void k_agg2(const long long* __restrict__ src,
                       const long long* __restrict__ dst,
                       const float* __restrict__ dinv,
                       const float* __restrict__ h2,
                       float* __restrict__ out, int nE) {
    int e = blockIdx.x * blockDim.x + threadIdx.x;
    if (e >= nE) return;
    int ep = e + 8192;
    if (ep < nE) {
        __builtin_prefetch(&src[ep], 0, 0);
        __builtin_prefetch(&dst[ep], 0, 0);
    }
    int s = (int)src[e], d = (int)dst[e];
    float norm = dinv[s] * dinv[d];
    const float2 v = *(const float2*)(h2 + (size_t)s * OUT_CH);
    atomicAdd(&out[(size_t)d * OUT_CH + 0], v.x * norm);
    atomicAdd(&out[(size_t)d * OUT_CH + 1], v.y * norm);
}

// ---------------------------------------------------------------------------
extern "C" void kernel_launch(void* const* d_in, const int* in_sizes, int n_in,
                              void* d_out, int out_size, void* d_ws, size_t ws_size,
                              hipStream_t stream) {
    const float*     x  = (const float*)d_in[0];
    const long long* ei = (const long long*)d_in[1];   // int64 [2, E]
    // d_in[2] = edge_attr (unused by GCNConv)
    const float* W1 = (const float*)d_in[3];
    const float* b1 = (const float*)d_in[4];
    const float* W2 = (const float*)d_in[5];
    const float* b2 = (const float*)d_in[6];

    const int n  = in_sizes[0] / IN_CH;
    const int nE = in_sizes[1] / 2;
    const long long* src = ei;
    const long long* dst = ei + nE;

    // workspace layout (f32 words): [deg/dinv: n][agg1: n*16][h1: n*16][h2: n*2]
    float* ws   = (float*)d_ws;
    float* dinv = ws;
    float* agg1 = ws + n;
    float* h1   = agg1 + (size_t)n * HID;
    float* h2   = h1   + (size_t)n * HID;
    float* out  = (float*)d_out;

    const size_t nz = (size_t)n * (1 + HID);           // deg + agg1
    k_zero  <<<(unsigned)((nz + 255) / 256), 256, 0, stream>>>(ws, nz);
    k_degree<<<(nE + 255) / 256, 256, 0, stream>>>(dst, (unsigned*)dinv, nE);
    k_dinv  <<<(n + 255) / 256, 256, 0, stream>>>((unsigned*)dinv, n);

    const int tiles = (n + 15) / 16;                   // one wave per 16-node tile
    k_gemm1_wmma<<<(tiles + 7) / 8, 256, 0, stream>>>(x, W1, h1, n);

    const long long t1 = (long long)nE * 4;            // 4 lanes / edge
    k_agg1<<<(unsigned)((t1 + 255) / 256), 256, 0, stream>>>(src, dst, dinv, h1, agg1, nE);

    k_finish1_gemm2<<<(n + 255) / 256, 256, 0, stream>>>(agg1, h1, dinv, b1, W2, b2, h2, out, n);

    k_agg2<<<(nE + 255) / 256, 256, 0, stream>>>(src, dst, dinv, h2, out, nE);
}
